// GraphColorizer_64141041598551
// MI455X (gfx1250) — matrix operand order repaired and checked by reading, not compiled, not run on hardware
//
#include <hip/hip_runtime.h>
#include <hip/hip_bf16.h>
#include <math.h>

typedef __bf16 bf16_t;
typedef bf16_t v16bf __attribute__((ext_vector_type(16)));
typedef float  v8f   __attribute__((ext_vector_type(8)));
typedef unsigned short v16us __attribute__((ext_vector_type(16)));

#define GAS __attribute__((address_space(1)))

#define N_V    512
#define DEG    8
#define EMB    512
#define NCOL   256
#define NCLS   257
#define HEADS  2
#define HID    512
#define INDIM  768
#define KTILES 24                       // 768 / 32  (front GEMM K tiles)
#define NTILES 32                       // 512 / 16  (N tiles)
#define KT2    32                       // 1024 / 32 (Wout GEMM K tiles)
#define COLTILES (HEADS * NTILES)       // 64 column tiles total
#define NTHR   512
#define NWAVES (NTHR / 32)              // 16 wave32s
#define COLS_PER_WAVE (COLTILES / NWAVES)  // 4
#define NEGV   (-1.0e9f)

// workspace byte offsets
#define WS_WBF16_BYTES   (HEADS * NTILES * KTILES * 512 * 2)   // 1,572,864 B
#define WS_WOUT_OFF      WS_WBF16_BYTES
#define WS_WOUT_BYTES    (NTILES * KT2 * 512 * 2)              // 1,048,576 B
#define WS_EMB_OFF       (WS_WOUT_OFF + WS_WOUT_BYTES)
#define WS_EMB_BYTES     (N_V * EMB * 4)                       // 1,048,576 B
#define WS_NEWEMB_OFF    (WS_EMB_OFF + WS_EMB_BYTES)           // 8*512*4 B scratch

__device__ __forceinline__ unsigned short f32_to_bf16u(float f) {
  unsigned int u = __float_as_uint(f);
  u += 0x7FFFu + ((u >> 16) & 1u);                             // round-to-nearest-even
  return (unsigned short)(u >> 16);
}

// Load one 32B WMMA fragment chunk through an explicit global-AS pointer so the
// backend emits global_load_b128 (LOADcnt only) instead of flat_load_b128
// (LOADcnt+DScnt, which forces joint full-drain waits against the LDS A-loads).
__device__ __forceinline__ v16bf gload_frag(const unsigned short* p) {
  const GAS unsigned short* gp = (const GAS unsigned short*)p;
  return *(const GAS v16bf*)gp;
}

// W [H][768][512] f32 -> bf16, pre-swizzled into WMMA B-fragment lane order:
// frag(col,kt) with col = head*32+nt; element (lane*16+e) =
//   W[head][kt*32+(e>=8?16:0)+(lane>=16?8:0)+e%8][nt*16+lane%16]
__global__ void prep_weights_bf16(const float* __restrict__ W,
                                  unsigned short* __restrict__ wsW) {
  int idx = blockIdx.x * blockDim.x + threadIdx.x;
  if (idx >= COLTILES * KTILES * 512) return;
  int p = idx & 511;                // lane*16 + e
  int f = idx >> 9;                 // col*KTILES + kt
  int kt = f % KTILES;
  int col = f / KTILES;             // head*32 + nt
  int nt = col & 31, h = col >> 5;
  int lane = p >> 4, e = p & 15;
  int k = kt * 32 + ((e >> 3) << 4) + ((lane >> 4) << 3) + (e & 7);
  int n = nt * 16 + (lane & 15);
  wsW[idx] = f32_to_bf16u(W[(h * INDIM + k) * HID + n]);
}

// Wout [1024][512] f32 -> bf16 B-fragments: frag(nt,kt) at (nt*KT2+kt)*512
__global__ void prep_wout_bf16(const float* __restrict__ Wout,
                               unsigned short* __restrict__ wsWout) {
  int idx = blockIdx.x * blockDim.x + threadIdx.x;
  if (idx >= NTILES * KT2 * 512) return;
  int p = idx & 511;
  int f = idx >> 9;                 // nt*KT2 + kt
  int kt = f & (KT2 - 1);
  int nt = f >> 5;
  int lane = p >> 4, e = p & 15;
  int k = kt * 32 + ((e >> 3) << 4) + ((lane >> 4) << 3) + (e & 7);
  int n = nt * 16 + (lane & 15);
  wsWout[idx] = f32_to_bf16u(Wout[k * EMB + n]);
}

__global__ void __launch_bounds__(NTHR, 1)
graph_colorize(const int* __restrict__ adj,
               const int* __restrict__ vord,
               const float* __restrict__ a_att,
               const float* __restrict__ b_out,
               const float* __restrict__ fc1w, const float* __restrict__ fc1b,
               const float* __restrict__ fc2w, const float* __restrict__ fc2b,
               const float* __restrict__ fc3w, const float* __restrict__ fc3b,
               const unsigned short* __restrict__ wsW,
               const unsigned short* __restrict__ wsWout,
               float* __restrict__ emb,
               float* __restrict__ newemb,
               float* __restrict__ out) {
  __shared__ __align__(32) unsigned short ldsA[KTILES * 512]; // 24.5 KB A-fragments (bf16)
  __shared__ float ldsH[HEADS * 9 * HID];                     // 36 KB H = X@W rows 0..8
  __shared__ int   colS[N_V];                                 // 2 KB colors
  __shared__ float e0S[HEADS];
  __shared__ float ejS[HEADS * 9];
  __shared__ float attS[HEADS * 9];
  __shared__ float redS[2];                                   // [max, sum]
  __shared__ int   nusedS;

  const int tid = threadIdx.x;
  // overlays of the A-tile region (lifetimes disjoint with front GEMM reads):
  float* const fscratch = (float*)ldsA;                 // classify activations
  unsigned short* const aggF = (unsigned short*)ldsA;   // Wout A-fragments (2 KB)

  // ---- one SGAT forward for vertex v; writes ELU output (512 f32) to dst ----
  auto sgat = [&](int v, float* dst) {
    // zero A tile (vectorized; rows 9..15 stay zero)
    for (int p = tid; p < (KTILES * 512) / 8; p += NTHR)
      ((uint4*)ldsA)[p] = make_uint4(0u, 0u, 0u, 0u);
    __syncthreads();
    // gather X = [emb | onehot(colors)] rows {v, adj[v]} straight into A-frag layout
    for (int p = tid; p < 9 * INDIM; p += NTHR) {
      int m = p / INDIM, k = p - m * INDIM;
      int vid = (m == 0) ? v : adj[v * DEG + (m - 1)];
      float val = (k < EMB) ? emb[vid * EMB + k]
                            : ((colS[vid] == (k - EMB)) ? 1.0f : 0.0f);
      int kt = k >> 5, w = k & 31, half = w >> 4, sub = w & 15;
      int lane = m + ((sub >= 8) ? 16 : 0);
      int e = half * 8 + (sub & 7);
      ldsA[kt * 512 + lane * 16 + e] = f32_to_bf16u(val);
    }
    __syncthreads();
    // H[h][16][512] = X @ W[h]  (v_wmma_f32_16x16x32_bf16, f32 accumulate)
    {
      const int wave = tid >> 5, lane = tid & 31;
      // single per-wave base pointer; columns/kt become constant offsets so the
      // backend uses global_load_b128 base+imm addressing
      const unsigned short* wbase = wsW + wave * (KTILES * 512) + lane * 16;
      v8f acc[COLS_PER_WAVE];
      for (int c = 0; c < COLS_PER_WAVE; ++c)
        for (int i = 0; i < 8; ++i) acc[c][i] = 0.0f;
      for (int kt = 0; kt < KTILES; ++kt) {
        v16bf a = *(const v16bf*)&ldsA[kt * 512 + lane * 16];
#pragma unroll
        for (int c = 0; c < COLS_PER_WAVE; ++c) {
          v16bf b = gload_frag(wbase + c * (16 * KTILES * 512) + kt * 512);
          acc[c] = __builtin_amdgcn_wmma_f32_16x16x32_bf16(
              false, a, false, b, (short)0, acc[c], false, false);
        }
      }
      const int rbase = (lane >> 4) << 3, nl = lane & 15;
#pragma unroll
      for (int c = 0; c < COLS_PER_WAVE; ++c) {
        int col = wave + (c << 4);
        int head = col >> 5, nt = col & 31;
#pragma unroll
        for (int r = 0; r < 8; ++r) {
          int row = rbase + r;
          if (row < 9) ldsH[(head * 9 + row) * HID + nt * 16 + nl] = acc[c][r];
        }
      }
    }
    __syncthreads();
    // attention logits: e0[h] = h0 . a1 ;  ej[h][m] = h_m . a2
    if (tid < HEADS) e0S[tid] = 0.0f;
    if (tid < HEADS * 9) ejS[tid] = 0.0f;
    __syncthreads();
    {
      int h = tid >> 8;                 // 0..1
      int n0 = (tid & 255) * 2;         // 2 columns per thread
      float pe0 = 0.0f, pej[9];
      for (int m = 0; m < 9; ++m) pej[m] = 0.0f;
      for (int d = 0; d < 2; ++d) {
        int n = n0 + d;
        float a1 = a_att[h * (2 * HID) + n];
        float a2 = a_att[h * (2 * HID) + HID + n];
        pe0 = fmaf(ldsH[(h * 9) * HID + n], a1, pe0);
        for (int m = 0; m < 9; ++m)
          pej[m] = fmaf(ldsH[(h * 9 + m) * HID + n], a2, pej[m]);
      }
      atomicAdd(&e0S[h], pe0);                 // ds_add_f32 reductions
      for (int m = 0; m < 9; ++m) atomicAdd(&ejS[h * 9 + m], pej[m]);
    }
    __syncthreads();
    if (tid < HEADS) {
      int h = tid;
      float e[9], mx = -3.4e38f;
      for (int m = 0; m < 9; ++m) {
        float t = e0S[h] + ejS[h * 9 + m];
        t = (t > 0.0f) ? t : 0.2f * t;         // LeakyReLU(0.2)
        e[m] = t; if (t > mx) mx = t;
      }
      float s = 0.0f;
      for (int m = 0; m < 9; ++m) { e[m] = expf(e[m] - mx); s += e[m]; }
      for (int m = 0; m < 9; ++m) attS[h * 9 + m] = e[m] / s;
    }
    __syncthreads();
    // agg (concat heads): pack bf16(agg[g]) DIRECTLY into the M=1 WMMA
    // A-fragment layout (aggF aliases the dead front A-tile). Row m=0 only:
    // position = kt*32 + (k%16>=8 ? 16:0) + (k%32/16)*8 + (k%8); bijective,
    // all 1024 slots written -> no zero fill of aggF needed.
    for (int g = tid; g < HEADS * HID; g += NTHR) {
      int h = g >> 9, n = g & (HID - 1);
      float s = 0.0f;
      for (int m = 0; m < 9; ++m)
        s = fmaf(attS[h * 9 + m], ldsH[(h * 9 + m) * HID + n], s);
      int kt = g >> 5, w = g & 31, half = w >> 4, sub = w & 15;
      aggF[kt * 32 + ((sub >= 8) ? 16 : 0) + half * 8 + (sub & 7)] = f32_to_bf16u(s);
    }
    __syncthreads();
    // dst = elu(agg @ Wout + b_out) as an M=1 WMMA GEMM over K=1024.
    // Only lanes 0 and 16 carry the A row: one 32B ds_load each per K-tile;
    // every other lane contributes a zero fragment.
    {
      const int wave = tid >> 5, lane = tid & 31;
      const bool row0 = ((lane & 15) == 0);
      const unsigned short* obase = wsWout + wave * (KT2 * 512) + lane * 16;
      v16us zu;
#pragma unroll
      for (int e = 0; e < 16; ++e) zu[e] = 0;
      const v16bf zfrag = __builtin_bit_cast(v16bf, zu);
      v8f acc[2];
      for (int c = 0; c < 2; ++c)
        for (int i = 0; i < 8; ++i) acc[c][i] = 0.0f;
      for (int kt = 0; kt < KT2; ++kt) {
        v16bf a = zfrag;
        if (row0) a = *(const v16bf*)&aggF[kt * 32 + ((lane >> 4) << 4)];
#pragma unroll
        for (int c = 0; c < 2; ++c) {
          v16bf b = gload_frag(obase + c * (16 * KT2 * 512) + kt * 512);
          acc[c] = __builtin_amdgcn_wmma_f32_16x16x32_bf16(
              false, a, false, b, (short)0, acc[c], false, false);
        }
      }
      if (lane < 16) {                  // D row 0 lives in VGPR0 of lanes 0..15
#pragma unroll
        for (int c = 0; c < 2; ++c) {
          int n = (wave + (c << 4)) * 16 + lane;
          float s = acc[c][0] + b_out[n];
          dst[n] = (s > 0.0f) ? s : (expf(s) - 1.0f);
        }
      }
    }
    __syncthreads();
  };

  // recompute emb for all 8 neighbors of v from the same snapshot, then commit
  auto update_neighbors = [&](int v) {
    for (int i = 0; i < DEG; ++i)
      sgat(adj[v * DEG + i], newemb + i * EMB);
    for (int p = tid; p < DEG * EMB; p += NTHR) {
      int i = p >> 9;
      emb[adj[v * DEG + i] * EMB + (p & (EMB - 1))] = newemb[p];
    }
    __syncthreads();
  };

  // 3-layer classifier + masked softmax + greedy color choice
  auto classify = [&](int v, int row) {
    float* z1 = fscratch;           // 512
    float* z2 = fscratch + 512;     // 400
    float* lg = fscratch + 912;     // 257
    {
      int j = tid;
      const float* ev = emb + v * EMB;
      float s = fc1b[j];
      for (int k = 0; k < EMB; ++k) s = fmaf(ev[k], fc1w[k * EMB + j], s);
      z1[j] = (s > 0.0f) ? s : 0.01f * s;
    }
    __syncthreads();
    if (tid < 400) {
      float s = fc2b[tid];
      for (int k = 0; k < EMB; ++k) s = fmaf(z1[k], fc2w[k * 400 + tid], s);
      z2[tid] = (s > 0.0f) ? s : 0.01f * s;
    }
    __syncthreads();
    if (tid < NCLS) {
      float s = fc3b[tid];
      for (int k = 0; k < 400; ++k) s = fmaf(z2[k], fc3w[k * NCLS + tid], s);
      bool irr = (tid >= nusedS) && (tid < NCOL);
      bool am = false;
      for (int d = 0; d < DEG; ++d) am |= (colS[adj[v * DEG + d]] == tid);
      lg[tid] = (irr || am) ? NEGV : s;
    }
    __syncthreads();
    if (tid == 0) {
      float mx = lg[0];
      for (int j = 1; j < NCLS; ++j) if (lg[j] > mx) mx = lg[j];
      redS[0] = mx;
    }
    __syncthreads();
    if (tid < NCLS) lg[tid] = expf(lg[tid] - redS[0]);
    __syncthreads();
    if (tid == 0) {
      float s = 0.0f;
      for (int j = 0; j < NCLS; ++j) s += lg[j];
      redS[1] = s;
    }
    __syncthreads();
    if (tid < NCLS) {
      float p = lg[tid] / redS[1];
      lg[tid] = p;
      out[(size_t)row * NCLS + tid] = p;
    }
    __syncthreads();
    if (tid == 0) {
      int amax = 0; float bv = lg[0];
      for (int j = 1; j < NCLS; ++j) if (lg[j] > bv) { bv = lg[j]; amax = j; }
      int isnew = (amax == NCOL) ? 1 : 0;
      int chosen = isnew ? ((nusedS < NCOL - 1) ? nusedS : (NCOL - 1)) : amax;
      nusedS += isnew;
      colS[v] = chosen;
    }
    __syncthreads();
  };

  // ---- init state (deterministic every call) ----
  for (int i = tid; i < N_V; i += NTHR) colS[i] = -1;
  if (tid == 0) nusedS = 0;
  for (int p = tid; p < N_V * EMB; p += NTHR) emb[p] = 0.0f;
  __syncthreads();
  int v0 = vord[0];
  if (tid == 0) colS[v0] = 0;
  __syncthreads();
  update_neighbors(v0);

  // ---- the 511-step sequential scan ----
  for (int t = 1; t < N_V; ++t) {
    int v = vord[t];
    sgat(v, emb + v * EMB);     // refresh v's embedding with current colors
    classify(v, t - 1);         // pick color, emit probs row
    update_neighbors(v);        // propagate v's new color to neighbors
  }

  // outputs: probs [511][257] (written above), colors [512], n_used
  for (int i = tid; i < N_V; i += NTHR)
    out[(size_t)(N_V - 1) * NCLS + i] = (float)colS[i];
  if (tid == 0) out[(size_t)(N_V - 1) * NCLS + N_V] = (float)nusedS;
}

extern "C" void kernel_launch(void* const* d_in, const int* in_sizes, int n_in,
                              void* d_out, int out_size, void* d_ws, size_t ws_size,
                              hipStream_t stream) {
  const int*   adj   = (const int*)d_in[0];
  const int*   vord  = (const int*)d_in[1];
  const float* W     = (const float*)d_in[2];
  const float* a_att = (const float*)d_in[3];
  const float* Wout  = (const float*)d_in[4];
  const float* b_out = (const float*)d_in[5];
  const float* fc1w  = (const float*)d_in[6];
  const float* fc1b  = (const float*)d_in[7];
  const float* fc2w  = (const float*)d_in[8];
  const float* fc2b  = (const float*)d_in[9];
  const float* fc3w  = (const float*)d_in[10];
  const float* fc3b  = (const float*)d_in[11];

  unsigned short* wsW    = (unsigned short*)d_ws;
  unsigned short* wsWout = (unsigned short*)((char*)d_ws + WS_WOUT_OFF);
  float* emb    = (float*)((char*)d_ws + WS_EMB_OFF);
  float* newemb = (float*)((char*)d_ws + WS_NEWEMB_OFF);

  int totW = COLTILES * KTILES * 512;
  int totO = NTILES * KT2 * 512;
  hipLaunchKernelGGL(prep_weights_bf16, dim3((totW + 255) / 256), dim3(256), 0,
                     stream, W, wsW);
  hipLaunchKernelGGL(prep_wout_bf16, dim3((totO + 255) / 256), dim3(256), 0,
                     stream, Wout, wsWout);
  hipLaunchKernelGGL(graph_colorize, dim3(1), dim3(NTHR), 0, stream,
                     adj, vord, a_att, b_out,
                     fc1w, fc1b, fc2w, fc2b, fc3w, fc3b,
                     wsW, wsWout, emb, newemb, (float*)d_out);
}
